// MELDDialogueGCN_25091198943369
// MI455X (gfx1250) — compile-verified, matrix operand order checked by reading, not run on hardware
//
#include <hip/hip_runtime.h>
#include <hip/hip_bf16.h>

typedef __attribute__((ext_vector_type(16))) _Float16 v16h;
typedef __attribute__((ext_vector_type(8)))  _Float16 v8h;
typedef __attribute__((ext_vector_type(8)))  float    v8f;

#define Tn 110
#define Bn 256
#define Nn (Tn*Bn)   // 28160

// ---------------- generic WMMA GEMM: C = A * Bt^T (+bias, relu), batched ----------------
// A:  [Mrows x K] f16 row-major (lda halves); rows >= Mrows redirected to zero row
// Bt: [nBtRows x K] f16 row-major (= B transposed); rows >= nBtRows redirected to zero row
// C:  [Mrows x 16*nTilesN]; f32 and/or f16 outputs
__global__ __launch_bounds__(256)
void k_wmma_gemm(const _Float16* __restrict__ A, int lda, long long sA,
                 const _Float16* __restrict__ Bt, int ldb, long long sB, int nBtRows,
                 float* __restrict__ C32, int ldc32, long long sC32,
                 _Float16* __restrict__ C16, int ldc16, long long sC16,
                 int Mrows, int nTilesN, int K,
                 const float* __restrict__ bias, int nBias, int relu,
                 const _Float16* __restrict__ zrow)
{
    const int wave = threadIdx.x >> 5;
    const int lane = threadIdx.x & 31;
    const int mTiles = (Mrows + 15) >> 4;
    const long long tile = (long long)blockIdx.x * 8 + wave;
    if (tile >= (long long)mTiles * nTilesN) return;
    const int tm = (int)(tile / nTilesN);
    const int tn = (int)(tile % nTilesN);

    const int half = lane >> 4;   // 0/1
    const int idx  = lane & 15;
    const int arow = tm * 16 + idx;
    const int brow = tn * 16 + idx;
    // loop-invariant masking by pointer redirection to a zero-filled row:
    // no exec-mask branching inside the K loop.
    const _Float16* pA = (arow < Mrows)
        ? (A + (long long)blockIdx.y * sA + (long long)arow * lda) : zrow;
    const _Float16* pB = (brow < nBtRows)
        ? (Bt + (long long)blockIdx.y * sB + (long long)brow * ldb) : zrow;

    v8f acc = {0.f,0.f,0.f,0.f,0.f,0.f,0.f,0.f};
    for (int k0 = 0; k0 < K; k0 += 32) {
        const int kA = k0 + half * 8;
        v8h a_lo = *(const v8h*)(pA + kA);
        v8h a_hi = *(const v8h*)(pA + kA + 16);
        v8h b_lo = *(const v8h*)(pB + kA);
        v8h b_hi = *(const v8h*)(pB + kA + 16);
        v16h a, b;
#pragma unroll
        for (int i = 0; i < 8; ++i) {
            a[i] = a_lo[i]; a[i+8] = a_hi[i];
            b[i] = b_lo[i]; b[i+8] = b_hi[i];
        }
        acc = __builtin_amdgcn_wmma_f32_16x16x32_f16(false, a, false, b,
                                                     (short)0, acc, false, false);
    }
    const int col = tn * 16 + idx;
    const float bval = (bias != nullptr && col < nBias) ? bias[col] : 0.0f;
#pragma unroll
    for (int v = 0; v < 8; ++v) {
        const int row = tm * 16 + v + half * 8;
        if (row < Mrows) {
            float r = acc[v] + bval;
            if (relu) r = fmaxf(r, 0.0f);
            if (C32) C32[(long long)blockIdx.y * sC32 + (long long)row * ldc32 + col] = r;
            if (C16) C16[(long long)blockIdx.y * sC16 + (long long)row * ldc16 + col] = (_Float16)r;
        }
    }
}

// ---------------- helper / prep kernels ----------------
__global__ __launch_bounds__(256) void k_zero16(uint4* p, long long n) {
    long long i = (long long)blockIdx.x * 256 + threadIdx.x;
    if (i < n) { uint4 z; z.x=0; z.y=0; z.z=0; z.w=0; p[i] = z; }
}

// fused input: A1[n, c] (K padded 300->320), n = b*T + t, c = concat(text,acou,visu)
__global__ __launch_bounds__(256)
void k_pack(const float* __restrict__ tf, const float* __restrict__ af,
            const float* __restrict__ vf, _Float16* __restrict__ A1, long long total) {
    long long id = (long long)blockIdx.x * 256 + threadIdx.x;
    if (id >= total) return;
    int c = (int)(id % 320); long long n = id / 320;
    int t = (int)(n % Tn); int b = (int)(n / Tn);
    long long base = ((long long)t * Bn + b) * 100;
    float v = 0.f;
    if (c < 100)      v = tf[base + c];
    else if (c < 200) v = af[base + c - 100];
    else if (c < 300) v = vf[base + c - 200];
    A1[id] = (_Float16)v;
}

// Wrels[r,f,h] = sum_nb comp[r,nb]*basis[nb,f,h]
__global__ __launch_bounds__(256)
void k_wrels(const float* __restrict__ comp, const float* __restrict__ basis,
             float* __restrict__ W) {
    int id = blockIdx.x * 256 + threadIdx.x;
    if (id >= 8*200*64) return;
    int r = id / (200*64); int rem = id % (200*64);
    int f = rem / 64; int h = rem % 64;
    float s = 0.f;
    for (int nb = 0; nb < 30; ++nb)
        s += comp[r*30 + nb] * basis[((long long)nb*200 + f)*64 + h];
    W[id] = s;
}

// Bt2[c, k]: c<512 -> Wrels[r=c/64][k][h=c%64], c>=512 -> root[k][c-512]; K pad 200->224
__global__ __launch_bounds__(256)
void k_bt2(const float* __restrict__ Wrels, const float* __restrict__ root,
           _Float16* __restrict__ Bt) {
    int id = blockIdx.x * 256 + threadIdx.x;
    if (id >= 576*224) return;
    int c = id / 224, k = id % 224;
    float v = 0.f;
    if (k < 200) {
        if (c < 512) v = Wrels[((c >> 6) * 200 + k) * 64 + (c & 63)];
        else         v = root[k * 64 + (c - 512)];
    }
    Bt[id] = (_Float16)v;
}

// generic weight transpose+pad: Bt[n,k] = (k<kreal) ? W[k*ncols+n] : 0
__global__ __launch_bounds__(256)
void k_wt(const float* __restrict__ W, _Float16* __restrict__ Bt,
          int ncols, int kreal, int kpad, int total) {
    int id = blockIdx.x * 256 + threadIdx.x;
    if (id >= total) return;
    int n = id / kpad, k = id % kpad;
    Bt[id] = (_Float16)((k < kreal) ? W[k * ncols + n] : 0.0f);
}

// Bt5[n,k] = k<64 ? Wrel[k,n] : Wroot2[k-64,n]   (64 x 128)
__global__ __launch_bounds__(256)
void k_bt5(const float* __restrict__ Wrel, const float* __restrict__ Wroot2,
           _Float16* __restrict__ Bt) {
    int id = blockIdx.x * 256 + threadIdx.x;
    if (id >= 64*128) return;
    int n = id / 128, k = id % 128;
    float v = (k < 64) ? Wrel[k*64 + n] : Wroot2[(k-64)*64 + n];
    Bt[id] = (_Float16)v;
}

// RGCN per-relation mean aggregation + root term; one 64-thread block per node
__global__ __launch_bounds__(64)
void k_rgcn(const float* __restrict__ xr, const int* __restrict__ speakers,
            const float* __restrict__ b_rgcn, float* __restrict__ h1,
            _Float16* __restrict__ A5) {
    int node = blockIdx.x;
    int b = node / Tn, i = node % Tn;
    int j0 = (i - 10 > 0) ? i - 10 : 0;
    int j1 = (i + 10 < Tn - 1) ? i + 10 : Tn - 1;
    int si = speakers[i * Bn + b];
    __shared__ float inv[8];
    if (threadIdx.x < 8) {
        int r = threadIdx.x, cnt = 0;
        for (int j = j0; j <= j1; ++j) {
            int et = speakers[j * Bn + b] * 4 + si * 2 + (j >= i ? 1 : 0);
            cnt += (et == r) ? 1 : 0;
        }
        inv[r] = 1.0f / (float)(cnt > 1 ? cnt : 1);
    }
    __syncthreads();
    int h = threadIdx.x;
    float acc = 0.f;
    for (int j = j0; j <= j1; ++j) {
        int et = speakers[j * Bn + b] * 4 + si * 2 + (j >= i ? 1 : 0);
        acc += inv[et] * xr[((long long)(b * Tn + j)) * 576 + et * 64 + h];
    }
    float v = acc + xr[(long long)node * 576 + 512 + h] + b_rgcn[h];
    h1[(long long)node * 64 + h] = v;
    A5[(long long)node * 128 + 64 + h] = (_Float16)v;   // h1 half of [agg|h1]
}

// GraphConv sum aggregation over the same window
__global__ __launch_bounds__(256)
void k_gconv(const float* __restrict__ h1, _Float16* __restrict__ A5, int total) {
    int id = blockIdx.x * 256 + threadIdx.x;
    if (id >= total) return;
    int node = id >> 6, h = id & 63;
    int b = node / Tn, i = node % Tn;
    int j0 = (i - 10 > 0) ? i - 10 : 0;
    int j1 = (i + 10 < Tn - 1) ? i + 10 : Tn - 1;
    float s = 0.f;
    for (int j = j0; j <= j1; ++j) s += h1[((long long)(b * Tn + j)) * 64 + h];
    A5[(long long)node * 128 + h] = (_Float16)s;
}

// tanh + softmax over s (110 valid of 112 cols), write f16 alpha padded to 128
__global__ __launch_bounds__(128)
void k_softmax(const float* __restrict__ scores, _Float16* __restrict__ alpha) {
    int row = blockIdx.x;                 // b*T + t
    const float* x = scores + (long long)row * 112;
    int s = threadIdx.x;
    __shared__ float red[128];
    float v = (s < Tn) ? tanhf(x[s]) : -1e30f;
    red[s] = v; __syncthreads();
    for (int off = 64; off > 0; off >>= 1) {
        if (s < off) red[s] = fmaxf(red[s], red[s + off]);
        __syncthreads();
    }
    float mx = red[0]; __syncthreads();
    float e = (s < Tn) ? __expf(v - mx) : 0.f;
    red[s] = e; __syncthreads();
    for (int off = 64; off > 0; off >>= 1) {
        if (s < off) red[s] += red[s + off];
        __syncthreads();
    }
    float sum = red[0];
    alpha[(long long)row * 128 + s] = (_Float16)(e / sum);
}

// per-dialogue transpose of M for the alpha @ M GEMM: Mt[b, d, s] = M[b*T+s, d]
__global__ __launch_bounds__(256)
void k_mt(const _Float16* __restrict__ M, _Float16* __restrict__ Mt, long long total) {
    long long id = (long long)blockIdx.x * 256 + threadIdx.x;
    if (id >= total) return;
    int s = (int)(id % 128); long long rem = id / 128;
    int d = (int)(rem % 264); int b = (int)(rem / 264);
    _Float16 v = (_Float16)0.f;
    if (s < Tn) v = M[((long long)(b * Tn + s)) * 288 + d];
    Mt[id] = v;
}

// log_softmax over 7 classes, one thread per node
__global__ __launch_bounds__(256)
void k_logsm(const float* __restrict__ logits, float* __restrict__ out, int total) {
    int id = blockIdx.x * 256 + threadIdx.x;
    if (id >= total) return;
    const float* x = logits + (long long)id * 16;
    float mx = x[0];
    for (int c = 1; c < 7; ++c) mx = fmaxf(mx, x[c]);
    float s = 0.f;
    for (int c = 0; c < 7; ++c) s += __expf(x[c] - mx);
    float l = __logf(s) + mx;
    for (int c = 0; c < 7; ++c) out[(long long)id * 7 + c] = x[c] - l;
}

extern "C" void kernel_launch(void* const* d_in, const int* in_sizes, int n_in,
                              void* d_out, int out_size, void* d_ws, size_t ws_size,
                              hipStream_t stream) {
    const float* textf  = (const float*)d_in[0];
    const float* acouf  = (const float*)d_in[1];
    const float* visuf  = (const float*)d_in[2];
    const int*   spk    = (const int*)d_in[3];
    const float* Wf     = (const float*)d_in[4];
    const float* bf     = (const float*)d_in[5];
    const float* basis  = (const float*)d_in[6];
    const float* comp   = (const float*)d_in[7];
    const float* root   = (const float*)d_in[8];
    const float* b_rgcn = (const float*)d_in[9];
    const float* Wrel   = (const float*)d_in[10];
    const float* brel   = (const float*)d_in[11];
    const float* Wroot2 = (const float*)d_in[12];
    const float* Wma    = (const float*)d_in[13];
    const float* bma    = (const float*)d_in[14];
    const float* Wl     = (const float*)d_in[15];
    const float* bl     = (const float*)d_in[16];
    const float* Ws     = (const float*)d_in[17];
    const float* bs     = (const float*)d_in[18];
    float* out = (float*)d_out;

    // ---- workspace layout (256B aligned) ----
    char* base = (char*)d_ws;
    size_t off = 0;
    auto alloc = [&](size_t bytes) { size_t o = off; off += (bytes + 255) & ~(size_t)255; return o; };
    _Float16* zrow   = (_Float16*)(base + alloc((size_t)512 * 2));        // zero row for masked lanes
    _Float16* A1     = (_Float16*)(base + alloc((size_t)Nn * 320 * 2));
    _Float16* Bt1    = (_Float16*)(base + alloc((size_t)200 * 320 * 2));
    float*    Wrels  = (float*)   (base + alloc((size_t)8 * 200 * 64 * 4));
    _Float16* Bt2    = (_Float16*)(base + alloc((size_t)576 * 224 * 2));
    float*    xr     = (float*)   (base + alloc((size_t)Nn * 576 * 4));
    _Float16* Mmat   = (_Float16*)(base + alloc((size_t)Nn * 288 * 2));
    float*    h1     = (float*)   (base + alloc((size_t)Nn * 64 * 4));
    _Float16* A5     = (_Float16*)(base + alloc((size_t)Nn * 128 * 2));
    _Float16* Bt5    = (_Float16*)(base + alloc((size_t)64 * 128 * 2));
    _Float16* Bt6    = (_Float16*)(base + alloc((size_t)264 * 288 * 2));
    _Float16* Xq     = (_Float16*)(base + alloc((size_t)Nn * 288 * 2));
    float*    scores = (float*)   (base + alloc((size_t)Bn * Tn * 112 * 4));
    _Float16* alpha  = (_Float16*)(base + alloc((size_t)Bn * Tn * 128 * 2));
    _Float16* Mt     = (_Float16*)(base + alloc((size_t)Bn * 264 * 128 * 2));
    _Float16* att    = (_Float16*)(base + alloc((size_t)Nn * 288 * 2));
    _Float16* Bt11   = (_Float16*)(base + alloc((size_t)64 * 288 * 2));
    _Float16* hidden = (_Float16*)(base + alloc((size_t)Nn * 64 * 2));
    _Float16* Bt12   = (_Float16*)(base + alloc((size_t)7 * 64 * 2));
    float*    logits = (float*)   (base + alloc((size_t)Nn * 16 * 4));
    (void)ws_size; (void)in_sizes; (void)n_in; (void)out_size;

    auto cdiv = [](long long a, long long b) { return (unsigned)((a + b - 1) / b); };
    auto gemm = [&](const _Float16* A, int lda, long long sA,
                    const _Float16* Bt, int ldb, long long sB, int nBtRows,
                    float* C32, int ldc32, long long sC32,
                    _Float16* C16, int ldc16, long long sC16,
                    int Mrows, int nTilesN, int K,
                    const float* bias, int nBias, int relu, int batches) {
        long long tiles = (long long)((Mrows + 15) / 16) * nTilesN;
        dim3 g(cdiv(tiles, 8), (unsigned)batches);
        k_wmma_gemm<<<g, 256, 0, stream>>>(A, lda, sA, Bt, ldb, sB, nBtRows,
                                           C32, ldc32, sC32, C16, ldc16, sC16,
                                           Mrows, nTilesN, K, bias, nBias, relu, zrow);
    };

    // zero: masked-lane row + buffers whose padding columns are never written by GEMMs
    {
        k_zero16<<<1, 256, 0, stream>>>((uint4*)zrow, 64);
        long long nM = (long long)Nn * 288 * 2 / 16;
        k_zero16<<<cdiv(nM, 256), 256, 0, stream>>>((uint4*)Mmat, nM);
        k_zero16<<<cdiv(nM, 256), 256, 0, stream>>>((uint4*)Xq,   nM);
        k_zero16<<<cdiv(nM, 256), 256, 0, stream>>>((uint4*)att,  nM);
    }

    // weight prep
    k_pack<<<cdiv((long long)Nn * 320, 256), 256, 0, stream>>>(textf, acouf, visuf, A1, (long long)Nn * 320);
    k_wt  <<<cdiv(200 * 320, 256), 256, 0, stream>>>(Wf,  Bt1, 200, 300, 320, 200 * 320);
    k_wrels<<<cdiv(8 * 200 * 64, 256), 256, 0, stream>>>(comp, basis, Wrels);
    k_bt2 <<<cdiv(576 * 224, 256), 256, 0, stream>>>(Wrels, root, Bt2);
    k_bt5 <<<cdiv(64 * 128, 256), 256, 0, stream>>>(Wrel, Wroot2, Bt5);
    k_wt  <<<cdiv(264 * 288, 256), 256, 0, stream>>>(Wma, Bt6, 264, 264, 288, 264 * 288);
    k_wt  <<<cdiv(64 * 288, 256), 256, 0, stream>>>(Wl, Bt11, 64, 264, 288, 64 * 288);
    k_wt  <<<cdiv(7 * 64, 256), 256, 0, stream>>>(Ws, Bt12, 7, 64, 64, 7 * 64);

    // G1: x = fused @ Wf + bf  -> M[:, 0:200] (f16), cols 200..207 get zeros
    gemm(A1, 320, 0, Bt1, 320, 0, 200,
         nullptr, 0, 0, Mmat, 288, 0,
         Nn, 13, 320, bf, 200, 0, 1);
    // G2: xr_ext = x @ [Wrels | root]  (f32, [N,576]); A read from M rows (cols 200..223 still zero)
    gemm(Mmat, 288, 0, Bt2, 224, 0, 576,
         xr, 576, 0, nullptr, 0, 0,
         Nn, 36, 224, nullptr, 0, 0, 1);
    // G3: RGCN per-relation mean + root + bias -> h1 (f32) and A5[:,64:128] (f16)
    k_rgcn<<<Nn, 64, 0, stream>>>(xr, spk, b_rgcn, h1, A5);
    // G4: GraphConv sum aggregation -> A5[:,0:64]
    k_gconv<<<cdiv((long long)Nn * 64, 256), 256, 0, stream>>>(h1, A5, Nn * 64);
    // G5: h2 = [agg|h1] @ [Wrel;Wroot2] + brel -> M[:, 200:264]
    gemm(A5, 128, 0, Bt5, 128, 0, 64,
         nullptr, 0, 0, Mmat + 200, 288, 0,
         Nn, 4, 128, brel, 64, 0, 1);
    // G6: Xq = M @ Wma + bma (f16)
    gemm(Mmat, 288, 0, Bt6, 288, 0, 264,
         nullptr, 0, 0, Xq, 288, 0,
         Nn, 17, 288, bma, 264, 0, 1);
    // G7 (batched over 256 dialogues): scores = Xq_b @ M_b^T (f32 [110 x 112])
    gemm(Xq, 288, (long long)Tn * 288, Mmat, 288, (long long)Tn * 288, Tn,
         scores, 112, (long long)Tn * 112, nullptr, 0, 0,
         Tn, 7, 288, nullptr, 0, 0, Bn);
    // G8: tanh + softmax -> alpha (f16, K padded to 128)
    k_softmax<<<Bn * Tn, 128, 0, stream>>>(scores, alpha);
    // G9: per-dialogue transposed M
    k_mt<<<cdiv((long long)Bn * 264 * 128, 256), 256, 0, stream>>>(Mmat, Mt, (long long)Bn * 264 * 128);
    // G10 (batched): att = alpha_b @ M_b (f16, node-major [N,288])
    gemm(alpha, 128, (long long)Tn * 128, Mt, 128, (long long)264 * 128, 264,
         nullptr, 0, 0, att, 288, (long long)Tn * 288,
         Tn, 17, 128, nullptr, 0, 0, Bn);
    // G11: hidden = relu(att @ Wl + bl) (f16)
    gemm(att, 288, 0, Bt11, 288, 0, 64,
         nullptr, 0, 0, hidden, 64, 0,
         Nn, 4, 288, bl, 64, 1, 1);
    // G12: logits = hidden @ Ws + bs (f32, [N,16])
    gemm(hidden, 64, 0, Bt12, 64, 0, 7,
         logits, 16, 0, nullptr, 0, 0,
         Nn, 1, 64, bs, 7, 0, 1);
    // G13: log_softmax -> d_out [N,7]
    k_logsm<<<cdiv(Nn, 256), 256, 0, stream>>>(logits, out, Nn);
}
